// Fp8Einsum_19808389169605
// MI455X (gfx1250) — 8-bit matrix operand order repaired and checked by reading, compiled, not run on hardware
//
#include <hip/hip_runtime.h>
#include <hip/hip_bf16.h>
#include <stdint.h>


typedef __attribute__((ext_vector_type(16))) int   v16i;
typedef __attribute__((ext_vector_type(8)))  float v8f;

#define D_DIM   256
#define F_DIM   256
#define TILE_M  64
#define TILE_N  128
#define LDA     272
#define LDB     272
#define E4M3MAX 448.0f

#if defined(__gfx1250__) && __has_builtin(__builtin_amdgcn_cvt_pk_fp8_f32)
#define HAVE_HW_FP8_CVT 1
#endif

__device__ __forceinline__ unsigned int f32_to_e4m3_sw(float x) {
    unsigned int bits = __float_as_uint(x);
    unsigned int sign = (bits >> 24) & 0x80u;
    bits &= 0x7FFFFFFFu;
    float a = __uint_as_float(bits);
    a = fminf(a, E4M3MAX);
    bits = __float_as_uint(a);
    unsigned int exp32 = bits >> 23;
    unsigned int code;
    if (exp32 >= 121u) {
        unsigned int lsb = (bits >> 20) & 1u;
        code = ((bits + 0x7FFFFu + lsb) >> 20) - 960u;
    } else {
        code = (unsigned int)rintf(a * 512.0f);
    }
    return sign | code;
}

__device__ __forceinline__ float clamp448(float t) {
    return fminf(fmaxf(t, -E4M3MAX), E4M3MAX);
}

__device__ __forceinline__ unsigned int quant4(float4 v, float inv) {
    float a = clamp448(v.x * inv);
    float b = clamp448(v.y * inv);
    float c = clamp448(v.z * inv);
    float d = clamp448(v.w * inv);
#ifdef HAVE_HW_FP8_CVT
    int p = 0;
    p = __builtin_amdgcn_cvt_pk_fp8_f32(a, b, p, false);
    p = __builtin_amdgcn_cvt_pk_fp8_f32(c, d, p, true);
    return (unsigned int)p;
#else
    unsigned int b0 = f32_to_e4m3_sw(a);
    unsigned int b1 = f32_to_e4m3_sw(b);
    unsigned int b2 = f32_to_e4m3_sw(c);
    unsigned int b3 = f32_to_e4m3_sw(d);
    return b0 | (b1 << 8) | (b2 << 16) | (b3 << 24);
#endif
}


__global__ void init_ws_kernel(unsigned int* w) {
    if (threadIdx.x < 8) w[threadIdx.x] = 0u;
}

__global__ void amax_kernel(const float4* __restrict__ x, int n4,
                            unsigned int* __restrict__ out_bits) {
    float m = 0.0f;
    int stride = gridDim.x * blockDim.x;
    for (int i = blockIdx.x * blockDim.x + threadIdx.x; i < n4; i += stride) {
        float4 v = x[i];
        m = fmaxf(m, fmaxf(fmaxf(fabsf(v.x), fabsf(v.y)),
                           fmaxf(fabsf(v.z), fabsf(v.w))));
    }
    #pragma unroll
    for (int off = 16; off > 0; off >>= 1)
        m = fmaxf(m, __shfl_xor(m, off, 32));
    if ((threadIdx.x & 31) == 0)
        atomicMax(out_bits, __float_as_uint(m));
}

__global__ void scale_kernel(const unsigned int* __restrict__ bits,
                             const float* __restrict__ in_scale,
                             const float* __restrict__ k_scale,
                             float* __restrict__ wsf) {
    float ax = __uint_as_float(bits[0]);
    float ak = __uint_as_float(bits[1]);
    float sx = (ax > 0.0f && isfinite(ax)) ? (ax / E4M3MAX) : in_scale[0];
    float sk = (ak > 0.0f && isfinite(ak)) ? (ak / E4M3MAX) : k_scale[0];
    wsf[2] = 1.0f / sx;
    wsf[3] = 1.0f / sk;
    wsf[4] = sx * sk;
}

__global__ void quantk_kernel(const float* __restrict__ k,
                              const float* __restrict__ wsf,
                              unsigned int* __restrict__ kq) {
    float inv = wsf[3];
    int idx = blockIdx.x * blockDim.x + threadIdx.x;
    int f  = idx >> 6;
    int dq = idx & 63;
    int d0 = dq * 4;
    float4 v;
    v.x = k[(d0 + 0) * F_DIM + f];
    v.y = k[(d0 + 1) * F_DIM + f];
    v.z = k[(d0 + 2) * F_DIM + f];
    v.w = k[(d0 + 3) * F_DIM + f];
    kq[f * (D_DIM / 4) + dq] = quant4(v, inv);
}

__global__ __launch_bounds__(256) void fp8_gemm_kernel(
        const float*         __restrict__ x,
        const unsigned char* __restrict__ kq,
        const float*         __restrict__ wsf,
        float*               __restrict__ y) {
    __shared__ alignas(16) unsigned char Alds[TILE_M * LDA];
    __shared__ alignas(16) unsigned char Blds[TILE_N * LDB];

    const int tid  = threadIdx.x;
    const int nb   = blockIdx.x & 1;
    const int mb   = blockIdx.x >> 1;
    const int row0 = mb * TILE_M;
    const int col0 = nb * TILE_N;

    const float inv_x     = wsf[2];
    const float out_scale = wsf[4];

    {
        const int n    = tid >> 1;
        const int half = (tid & 1) * 128;
        const unsigned char* src = kq + (size_t)(col0 + n) * D_DIM + half;
        unsigned char*       dst = Blds + n * LDB + half;
        #pragma unroll
        for (int i = 0; i < 8; ++i)
            *(uint4*)(dst + i * 16) = *(const uint4*)(src + i * 16);
    }

    {
        const int r = tid >> 2;
        const int q = (tid & 3) * 64;
        const float* srcx = x + (size_t)(row0 + r) * D_DIM + q;
        unsigned char* dst = Alds + r * LDA + q;
        #pragma unroll
        for (int i = 0; i < 4; ++i) {
            float4 a = *(const float4*)(srcx + i * 16);
            float4 b = *(const float4*)(srcx + i * 16 + 4);
            float4 c = *(const float4*)(srcx + i * 16 + 8);
            float4 d = *(const float4*)(srcx + i * 16 + 12);
            uint4 pk;
            pk.x = quant4(a, inv_x);
            pk.y = quant4(b, inv_x);
            pk.z = quant4(c, inv_x);
            pk.w = quant4(d, inv_x);
            *(uint4*)(dst + i * 16) = pk;
        }
    }
    __syncthreads();

    const int lane = tid & 31;
    const int wid  = tid >> 5;
    const int mt   = wid & 3;
    const int nh   = wid >> 2;
    const int r    = lane & 15;
    const int g    = lane >> 4;

    v16i Af[2];
    #pragma unroll
    for (int ks = 0; ks < 2; ++ks) {
        const unsigned char* ap = Alds + (mt * 16 + r) * LDA + ks * 128 + g * 16;
        #pragma unroll
        for (int p = 0; p < 4; ++p) {
            uint2 lo = *(const uint2*)(ap + (p >> 1) * 32 + (p & 1) * 8);
            uint2 hi = *(const uint2*)(ap + 64 + (p >> 1) * 32 + (p & 1) * 8);
            Af[ks][2 * p]     = (int)lo.x;
            Af[ks][2 * p + 1] = (int)lo.y;
            Af[ks][8 + 2 * p]     = (int)hi.x;
            Af[ks][8 + 2 * p + 1] = (int)hi.y;
        }
    }

    #pragma unroll
    for (int i = 0; i < 4; ++i) {
        const int nt = nh * 4 + i;
        v8f acc = {};
        #pragma unroll
        for (int ks = 0; ks < 2; ++ks) {
            v16i Bf;
            const unsigned char* bp =
                Blds + (nt * 16 + (lane & 15)) * LDB + ks * 128 + g * 16;
            #pragma unroll
            for (int q = 0; q < 4; ++q) {
                uint4 t = *(const uint4*)(bp + q * 32);
                Bf[4 * q]     = (int)t.x;
                Bf[4 * q + 1] = (int)t.y;
                Bf[4 * q + 2] = (int)t.z;
                Bf[4 * q + 3] = (int)t.w;
            }
            acc = __builtin_amdgcn_wmma_f32_16x16x128_fp8_fp8(
                      Af[ks], Bf, (short)0, acc, false, false);
        }
        float* yp = y + (size_t)(row0 + mt * 16 + g * 8) * F_DIM
                      + col0 + nt * 16 + (lane & 15);
        #pragma unroll
        for (int j = 0; j < 8; ++j)
            yp[(size_t)j * F_DIM] = acc[j] * out_scale;
    }
}

extern "C" void kernel_launch(void* const* d_in, const int* in_sizes, int n_in,
                              void* d_out, int out_size, void* d_ws, size_t ws_size,
                              hipStream_t stream) {
    const float* x        = (const float*)d_in[0];
    const float* k        = (const float*)d_in[1];
    const float* in_scale = (const float*)d_in[2];
    const float* k_scale  = (const float*)d_in[3];
    float*       y        = (float*)d_out;

    unsigned int* wsu = (unsigned int*)d_ws;
    float*        wsf = (float*)d_ws;
    unsigned char* kq = (unsigned char*)d_ws + 256;

    const int n_x = in_sizes[0];
    const int n_k = in_sizes[1];
    const int M   = n_x / D_DIM;

    hipLaunchKernelGGL(init_ws_kernel, dim3(1), dim3(32), 0, stream, wsu);
    hipLaunchKernelGGL(amax_kernel, dim3(4096), dim3(256), 0, stream,
                       (const float4*)x, n_x / 4, wsu + 0);
    hipLaunchKernelGGL(amax_kernel, dim3(64), dim3(256), 0, stream,
                       (const float4*)k, n_k / 4, wsu + 1);
    hipLaunchKernelGGL(scale_kernel, dim3(1), dim3(1), 0, stream,
                       wsu, in_scale, k_scale, wsf);
    hipLaunchKernelGGL(quantk_kernel, dim3(64), dim3(256), 0, stream,
                       k, wsf, (unsigned int*)kq);
    hipLaunchKernelGGL(fp8_gemm_kernel, dim3((M / TILE_M) * 2), dim3(256), 0,
                       stream, x, kq, wsf, y);
}
